// AttentiveGnLConv_19911468384990
// MI455X (gfx1250) — matrix operand order repaired and checked by reading, not compiled, not run on hardware
//
#include <hip/hip_runtime.h>
#include <hip/hip_bf16.h>
#include <math.h>

// Problem constants (match reference)
#define NN 50000
#define EE 600000

typedef __attribute__((ext_vector_type(16))) _Float16 v16h;
typedef __attribute__((ext_vector_type(8)))  _Float16 v8h;
typedef __attribute__((ext_vector_type(4)))  _Float16 v4h;
typedef __attribute__((ext_vector_type(2)))  _Float16 v2h;
typedef __attribute__((ext_vector_type(8)))  float    v8f;

// ---------- helpers ----------
__device__ __forceinline__ unsigned fmap(float x) {
  unsigned u = __float_as_uint(x);
  return (u & 0x80000000u) ? ~u : (u | 0x80000000u);
}
__device__ __forceinline__ float funmap(unsigned m) {
  unsigned u = (m & 0x80000000u) ? (m & 0x7FFFFFFFu) : ~m;
  return __uint_as_float(u);
}
__device__ __forceinline__ float sigmoidf_(float x) { return 1.0f / (1.0f + expf(-x)); }
__device__ __forceinline__ float eluf_(float x) { return x > 0.f ? x : expm1f(x); }

// ---------- WMMA GEMM: OUT[M,Nc] = act(X[M,K] @ W[K,Nc] + bias) ----------
// Requirements: K % 32 == 0 (compile-time), M % 16 == 0, Nc % 64 == 0, bias != null.
// grid = (M/16, Nc/64), block = 128 (4 waves); wave w computes one 16x16 tile.
// LDS row stride 40 halfs (80 B): 16-B aligned b128 accesses, conflict-spread.
template <int K>
__global__ __launch_bounds__(128) void gemm_wmma_kernel(
    const float* __restrict__ X, const float* __restrict__ W,
    const float* __restrict__ bias, float* __restrict__ OUT,
    int M, int Nc, int act)
{
  __shared__ _Float16 As[16][40];       // A tile 16x32 (row-major, K contiguous)
  __shared__ _Float16 Bt[4][16][40];    // B tile stored TRANSPOSED: Bt[w][n][k]
  const int tid  = threadIdx.x;
  const int lane = tid & 31;
  const int wv   = tid >> 5;
  const int g    = lane >> 4;   // lane group (0: lanes 0-15, 1: lanes 16-31)
  const int lm   = lane & 15;
  const int m0   = blockIdx.x * 16;
  const int n0   = blockIdx.y * 64 + wv * 16;

  const int ar  = tid >> 3;          // 0..15  (A row)
  const int ac4 = (tid & 7) << 2;    // 0..28  (A col, float4 granule)

  v8f acc = {};
  #pragma unroll
  for (int kk = 0; kk < K; kk += 32) {
    // ---- stage A tile: one float4 per thread (coalesced b128 loads) ----
    {
      const float4 f = *reinterpret_cast<const float4*>(
          X + (size_t)(m0 + ar) * K + kk + ac4);
      v4h h;
      h[0] = (_Float16)f.x; h[1] = (_Float16)f.y;
      h[2] = (_Float16)f.z; h[3] = (_Float16)f.w;
      *reinterpret_cast<v4h*>(&As[ar][ac4]) = h;   // 8-B aligned ds_store_b64
    }
    // ---- stage B tile transposed: row-pairs -> packed b32 LDS stores ----
    #pragma unroll
    for (int i = 0; i < 2; ++i) {
      int idx = lane + 32 * i;           // 0..63 -> 16 row-pairs x 4 col-granules
      int p   = idx >> 2;                // row pair: rows 2p, 2p+1
      int c4  = (idx & 3) << 2;          // col granule 0,4,8,12
      const float4 f0 = *reinterpret_cast<const float4*>(
          W + (size_t)(kk + 2 * p) * Nc + n0 + c4);
      const float4 f1 = *reinterpret_cast<const float4*>(
          W + (size_t)(kk + 2 * p + 1) * Nc + n0 + c4);
      v2h h0; h0[0] = (_Float16)f0.x; h0[1] = (_Float16)f1.x;
      v2h h1; h1[0] = (_Float16)f0.y; h1[1] = (_Float16)f1.y;
      v2h h2; h2[0] = (_Float16)f0.z; h2[1] = (_Float16)f1.z;
      v2h h3; h3[0] = (_Float16)f0.w; h3[1] = (_Float16)f1.w;
      *reinterpret_cast<v2h*>(&Bt[wv][c4 + 0][2 * p]) = h0;  // ds_store_b32
      *reinterpret_cast<v2h*>(&Bt[wv][c4 + 1][2 * p]) = h1;
      *reinterpret_cast<v2h*>(&Bt[wv][c4 + 2][2 * p]) = h2;
      *reinterpret_cast<v2h*>(&Bt[wv][c4 + 3][2 * p]) = h3;
    }
    __syncthreads();
    // ---- fragments: each = two contiguous 8-half runs -> 2x ds_load_b128 ----
    union { v16h v; v8h h[2]; } a, b;
    a.h[0] = *reinterpret_cast<const v8h*>(&As[lm][8 * g]);        // K: 8g..8g+7
    a.h[1] = *reinterpret_cast<const v8h*>(&As[lm][16 + 8 * g]);   // K: 16+8g..
    b.h[0] = *reinterpret_cast<const v8h*>(&Bt[wv][lm][8 * g]);        // K: 8g..8g+7
    b.h[1] = *reinterpret_cast<const v8h*>(&Bt[wv][lm][16 + 8 * g]);
    acc = __builtin_amdgcn_wmma_f32_16x16x32_f16(false, a.v, false, b.v,
                                                 (short)0, acc, false, false);
    __syncthreads();
  }
  const int col = n0 + lm;
  const float bb = bias[col];
  #pragma unroll
  for (int r = 0; r < 8; ++r) {
    int row = m0 + r + (g ? 8 : 0);
    float v = acc[r] + bb;
    if (act == 1) v = v > 0.f ? v : 0.01f * v;   // leaky_relu(0.01)
    OUT[(size_t)row * Nc + col] = v;
  }
}

// ---------- TransformerConv edge pass A: logits + per-(dst,head) max ----------
// one wave per edge; lane handles 4 of 128 channels; 8 lanes per head.
__global__ __launch_bounds__(256) void t_edge_logits_kernel(
    const float* __restrict__ q, const float* __restrict__ k,
    const float* __restrict__ ea, const float* __restrict__ te_w,
    const int* __restrict__ src, const int* __restrict__ dst,
    float* __restrict__ logits, unsigned* __restrict__ smax, int E)
{
  int lane = threadIdx.x & 31, wv = threadIdx.x >> 5;
  long eid = (long)blockIdx.x * 8 + wv;
  if (eid >= E) return;
  int s = src[eid], d = dst[eid];
  int ch = lane * 4, h = lane >> 3;
  float e0 = 0, e1 = 0, e2 = 0, e3 = 0;
  const float* eap = ea + eid * 16;
  #pragma unroll
  for (int j = 0; j < 16; ++j) {
    float aj = eap[j];
    const float* wr = te_w + j * 128 + ch;
    e0 += aj * wr[0]; e1 += aj * wr[1]; e2 += aj * wr[2]; e3 += aj * wr[3];
  }
  const float* kp = k + (long)s * 128 + ch;
  const float* qp = q + (long)d * 128 + ch;
  float part = qp[0] * (kp[0] + e0) + qp[1] * (kp[1] + e1)
             + qp[2] * (kp[2] + e2) + qp[3] * (kp[3] + e3);
  part += __shfl_xor(part, 1);
  part += __shfl_xor(part, 2);
  part += __shfl_xor(part, 4);
  float lg = part * 0.17677669529663689f;   // 1/sqrt(32)
  if ((lane & 7) == 0) {
    logits[eid * 4 + h] = lg;
    atomicMax(&smax[(long)d * 4 + h], fmap(lg));
  }
}

// ---------- pass B: exp + denominator ----------
__global__ void t_edge_expsum_kernel(float* __restrict__ lg,
                                     const unsigned* __restrict__ smax,
                                     float* __restrict__ den,
                                     const int* __restrict__ dst, int E)
{
  long idx = (long)blockIdx.x * blockDim.x + threadIdx.x;
  if (idx >= (long)E * 4) return;
  long eid = idx >> 2; int h = (int)(idx & 3);
  int d = dst[eid];
  float m = funmap(smax[(long)d * 4 + h]);
  float e = expf(lg[idx] - m);
  lg[idx] = e;
  atomicAdd(&den[(long)d * 4 + h], e);
}

// ---------- pass C: normalized message scatter ----------
__global__ __launch_bounds__(256) void t_edge_scatter_kernel(
    const float* __restrict__ v, const float* __restrict__ ea,
    const float* __restrict__ te_w, const int* __restrict__ src,
    const int* __restrict__ dst, const float* __restrict__ exb,
    const float* __restrict__ den, float* __restrict__ conv1, int E)
{
  int lane = threadIdx.x & 31, wv = threadIdx.x >> 5;
  long eid = (long)blockIdx.x * 8 + wv;
  if (eid >= E) return;
  int s = src[eid], d = dst[eid];
  int ch = lane * 4, h = lane >> 3;
  float e0 = 0, e1 = 0, e2 = 0, e3 = 0;
  const float* eap = ea + eid * 16;
  #pragma unroll
  for (int j = 0; j < 16; ++j) {
    float aj = eap[j];
    const float* wr = te_w + j * 128 + ch;
    e0 += aj * wr[0]; e1 += aj * wr[1]; e2 += aj * wr[2]; e3 += aj * wr[3];
  }
  float a = exb[eid * 4 + h] / (den[(long)d * 4 + h] + 1e-16f);
  const float* vp = v + (long)s * 128 + ch;
  float* cp = conv1 + (long)d * 128 + ch;
  atomicAdd(cp + 0, (vp[0] + e0) * a);
  atomicAdd(cp + 1, (vp[1] + e1) * a);
  atomicAdd(cp + 2, (vp[2] + e2) * a);
  atomicAdd(cp + 3, (vp[3] + e3) * a);
}

// ---------- GATv2 edge pass A ----------
__global__ __launch_bounds__(256) void g_edge_logits_kernel(
    const float* __restrict__ xl, const float* __restrict__ xr,
    const float* __restrict__ ea, const float* __restrict__ ge_w,
    const float* __restrict__ g_att, const int* __restrict__ src,
    const int* __restrict__ dst, float* __restrict__ logits,
    unsigned* __restrict__ smax, int E)
{
  int lane = threadIdx.x & 31, wv = threadIdx.x >> 5;
  long eid = (long)blockIdx.x * 8 + wv;
  if (eid >= E) return;
  int s = src[eid], d = dst[eid];
  int ch = lane * 2;
  float e0 = 0, e1 = 0;
  const float* eap = ea + eid * 16;
  #pragma unroll
  for (int j = 0; j < 16; ++j) {
    float aj = eap[j];
    e0 += aj * ge_w[j * 64 + ch];
    e1 += aj * ge_w[j * 64 + ch + 1];
  }
  float z0 = xl[(long)s * 64 + ch]     + xr[(long)d * 64 + ch]     + e0;
  float z1 = xl[(long)s * 64 + ch + 1] + xr[(long)d * 64 + ch + 1] + e1;
  z0 = z0 > 0.f ? z0 : 0.2f * z0;
  z1 = z1 > 0.f ? z1 : 0.2f * z1;
  float part = z0 * g_att[ch] + z1 * g_att[ch + 1];
  part += __shfl_xor(part, 1);
  part += __shfl_xor(part, 2);
  part += __shfl_xor(part, 4);
  part += __shfl_xor(part, 8);
  part += __shfl_xor(part, 16);
  if (lane == 0) {
    logits[eid] = part;
    atomicMax(&smax[d], fmap(part));
  }
}

__global__ void g_edge_expsum_kernel(float* __restrict__ lg,
                                     const unsigned* __restrict__ smax,
                                     float* __restrict__ den,
                                     const int* __restrict__ dst, int E)
{
  long eid = (long)blockIdx.x * blockDim.x + threadIdx.x;
  if (eid >= E) return;
  int d = dst[eid];
  float e = expf(lg[eid] - funmap(smax[d]));
  lg[eid] = e;
  atomicAdd(&den[d], e);
}

__global__ __launch_bounds__(256) void g_edge_scatter_kernel(
    const float* __restrict__ xl, const int* __restrict__ src,
    const int* __restrict__ dst, const float* __restrict__ exb,
    const float* __restrict__ den, float* __restrict__ conv2, int E)
{
  int lane = threadIdx.x & 31, wv = threadIdx.x >> 5;
  long eid = (long)blockIdx.x * 8 + wv;
  if (eid >= E) return;
  int s = src[eid], d = dst[eid];
  int ch = lane * 2;
  float a2 = exb[eid] / (den[d] + 1e-16f);
  atomicAdd(&conv2[(long)d * 64 + ch],     xl[(long)s * 64 + ch]     * a2);
  atomicAdd(&conv2[(long)d * 64 + ch + 1], xl[(long)s * 64 + ch + 1] * a2);
}

// ---------- elementwise ----------
__global__ void elu_add_kernel(float* __restrict__ a, const float* __restrict__ b, long n)
{
  long i = (long)blockIdx.x * blockDim.x + threadIdx.x;
  if (i >= n) return;
  a[i] = eluf_(a[i] + b[i]);
}

__global__ void elu_bias_kernel(float* __restrict__ a, const float* __restrict__ bias, long n, int hd)
{
  long i = (long)blockIdx.x * blockDim.x + threadIdx.x;
  if (i >= n) return;
  a[i] = eluf_(a[i] + bias[i % hd]);
}

__global__ void gru_relu_kernel(const float* __restrict__ gi, const float* __restrict__ gh,
                                float* __restrict__ hstate, int n, int hd)
{
  long idx = (long)blockIdx.x * blockDim.x + threadIdx.x;
  if (idx >= (long)n * hd) return;
  long row = idx / hd; int c = (int)(idx - row * hd);
  long b = row * 3 * hd;
  float ir = gi[b + c], iz = gi[b + hd + c], in_ = gi[b + 2 * hd + c];
  float hr = gh[b + c], hz = gh[b + hd + c], hn  = gh[b + 2 * hd + c];
  float r  = sigmoidf_(ir + hr);
  float z  = sigmoidf_(iz + hz);
  float nn = tanhf(in_ + r * hn);
  float out = (1.f - z) * nn + z * hstate[idx];
  hstate[idx] = fmaxf(out, 0.f);
}

// out = hprev + relu(gru(gi, gh, hprev))   (g_sum + x2, since g_sum == hprev)
__global__ void gru_final_kernel(const float* __restrict__ gi, const float* __restrict__ gh,
                                 const float* __restrict__ hprev, float* __restrict__ out,
                                 int n, int hd)
{
  long idx = (long)blockIdx.x * blockDim.x + threadIdx.x;
  if (idx >= (long)n * hd) return;
  long row = idx / hd; int c = (int)(idx - row * hd);
  long b = row * 3 * hd;
  float ir = gi[b + c], iz = gi[b + hd + c], in_ = gi[b + 2 * hd + c];
  float hr = gh[b + c], hz = gh[b + hd + c], hn  = gh[b + 2 * hd + c];
  float r  = sigmoidf_(ir + hr);
  float z  = sigmoidf_(iz + hz);
  float nn = tanhf(in_ + r * hn);
  float hp = hprev[idx];
  float x2 = fmaxf((1.f - z) * nn + z * hp, 0.f);
  out[idx] = hp + x2;
}

// ---------- host launch ----------
extern "C" void kernel_launch(void* const* d_in, const int* in_sizes, int n_in,
                              void* d_out, int out_size, void* d_ws, size_t ws_size,
                              hipStream_t stream)
{
  (void)in_sizes; (void)n_in; (void)out_size; (void)ws_size;
  const float* x       = (const float*)d_in[0];
  const float* ea      = (const float*)d_in[1];
  const float* tq_w    = (const float*)d_in[2];  const float* tq_b   = (const float*)d_in[3];
  const float* tk_w    = (const float*)d_in[4];  const float* tk_b   = (const float*)d_in[5];
  const float* tv_w    = (const float*)d_in[6];  const float* tv_b   = (const float*)d_in[7];
  const float* te_w    = (const float*)d_in[8];
  const float* ts_w    = (const float*)d_in[9];  const float* ts_b   = (const float*)d_in[10];
  const float* lin0_w  = (const float*)d_in[11]; const float* lin0_b = (const float*)d_in[12];
  const float* g1_wi   = (const float*)d_in[13]; const float* g1_wh  = (const float*)d_in[14];
  const float* g1_bi   = (const float*)d_in[15]; const float* g1_bh  = (const float*)d_in[16];
  const float* lin1_w  = (const float*)d_in[17]; const float* lin1_b = (const float*)d_in[18];
  const float* gl_w    = (const float*)d_in[19]; const float* gl_b   = (const float*)d_in[20];
  const float* gr_w    = (const float*)d_in[21]; const float* gr_b   = (const float*)d_in[22];
  const float* ge_w    = (const float*)d_in[23];
  const float* g_att   = (const float*)d_in[24]; const float* g_bias = (const float*)d_in[25];
  const float* g2_wi   = (const float*)d_in[26]; const float* g2_wh  = (const float*)d_in[27];
  const float* g2_bi   = (const float*)d_in[28]; const float* g2_bh  = (const float*)d_in[29];
  const int*   eidx    = (const int*)d_in[30];
  const int*   src     = eidx;
  const int*   dst     = eidx + EE;
  float* out = (float*)d_out;

  // ---- workspace layout (floats); total ~73.2M floats (~293 MB) ----
  float* ws = (float*)d_ws;
  const size_t NF  = (size_t)NN * 128;
  const size_t N64 = (size_t)NN * 64;
  float* q    = ws;                 // [N,128]
  float* k    = ws + NF;            // [N,128]
  float* v    = ws + 2 * NF;        // [N,128]
  float* sP   = ws + 3 * NF;        // root-skip projection [N,128]
  float* x1   = ws + 4 * NF;        // lin0 -> GRU1 state [N,128]
  float* cv1  = ws + 5 * NF;        // conv1 -> h [N,128]
  float* gh1  = ws + 6 * NF;        // [N,384] (spans 6NF..9NF)
  float* gi1  = ws;                 // reuse q/k/v region: [N,384]
  const size_t base2 = 9 * NF;
  float* x1o  = ws + base2;                 // g_sum / GRU2 state [N,64]
  float* xl   = ws + base2 + N64;           // [N,64]
  float* xr   = ws + base2 + 2 * N64;       // [N,64]
  float* cv2  = ws + base2 + 3 * N64;       // conv2 -> h2 [N,64]
  float* exb  = ws + base2 + 4 * N64;       // [E,4] logits/exp (reused [E] in phase 2)
  unsigned* smax = (unsigned*)(ws + base2 + 4 * N64 + (size_t)EE * 4); // [N,4]
  float* den  = ws + base2 + 4 * N64 + (size_t)EE * 4 + (size_t)NN * 4; // [N,4]
  float* gi2  = ws;                 // reuse region 0: [N,192]
  float* gh2  = ws + 6 * NF;        // reuse: [N,192]

  auto gemm = [&](const float* X, const float* W, const float* b, float* O,
                  int M, int K, int Nc, int act) {
    dim3 grid(M / 16, Nc / 64);
    if (K == 64)
      gemm_wmma_kernel<64><<<grid, 128, 0, stream>>>(X, W, b, O, M, Nc, act);
    else
      gemm_wmma_kernel<128><<<grid, 128, 0, stream>>>(X, W, b, O, M, Nc, act);
  };

  const int EB = (EE + 7) / 8;   // edge blocks (8 edges/block, 1 wave each)

  // ---- phase 1: node projections (WMMA) ----
  gemm(x, tq_w,   tq_b,   q,  NN, 64, 128, 0);
  gemm(x, tk_w,   tk_b,   k,  NN, 64, 128, 0);
  gemm(x, tv_w,   tv_b,   v,  NN, 64, 128, 0);
  gemm(x, ts_w,   ts_b,   sP, NN, 64, 128, 0);
  gemm(x, lin0_w, lin0_b, x1, NN, 64, 128, 1);   // leaky 0.01

  // zero accumulators / softmax state
  hipMemsetAsync(cv1,  0, NF * sizeof(float), stream);
  hipMemsetAsync(cv2,  0, N64 * sizeof(float), stream);
  hipMemsetAsync(smax, 0, (size_t)NN * 4 * sizeof(unsigned), stream);
  hipMemsetAsync(den,  0, (size_t)NN * 4 * sizeof(float), stream);

  // ---- TransformerConv edge passes ----
  t_edge_logits_kernel<<<EB, 256, 0, stream>>>(q, k, ea, te_w, src, dst, exb, smax, EE);
  t_edge_expsum_kernel<<<((size_t)EE * 4 + 255) / 256, 256, 0, stream>>>(exb, smax, den, dst, EE);
  t_edge_scatter_kernel<<<EB, 256, 0, stream>>>(v, ea, te_w, src, dst, exb, den, cv1, EE);
  elu_add_kernel<<<(NF + 255) / 256, 256, 0, stream>>>(cv1, sP, (long)NF);   // h = elu(conv1 + skip)

  // ---- GRU1 + lin1 ----
  gemm(cv1, g1_wi, g1_bi, gi1, NN, 128, 384, 0);
  gemm(x1,  g1_wh, g1_bh, gh1, NN, 128, 384, 0);
  gru_relu_kernel<<<(NF + 255) / 256, 256, 0, stream>>>(gi1, gh1, x1, NN, 128);
  gemm(x1, lin1_w, lin1_b, x1o, NN, 128, 64, 1);   // g_sum

  // ---- GATv2 ----
  gemm(x1o, gl_w, gl_b, xl, NN, 64, 64, 0);
  gemm(x1o, gr_w, gr_b, xr, NN, 64, 64, 0);
  hipMemsetAsync(smax, 0, (size_t)NN * sizeof(unsigned), stream);
  hipMemsetAsync(den,  0, (size_t)NN * sizeof(float), stream);
  g_edge_logits_kernel<<<EB, 256, 0, stream>>>(xl, xr, ea, ge_w, g_att, src, dst, exb, smax, EE);
  g_edge_expsum_kernel<<<(EE + 255) / 256, 256, 0, stream>>>(exb, smax, den, dst, EE);
  g_edge_scatter_kernel<<<EB, 256, 0, stream>>>(xl, src, dst, exb, den, cv2, EE);
  elu_bias_kernel<<<(N64 + 255) / 256, 256, 0, stream>>>(cv2, g_bias, (long)N64, 64);  // h2

  // ---- GRU2 + final add ----
  gemm(cv2, g2_wi, g2_bi, gi2, NN, 64, 192, 0);
  gemm(x1o, g2_wh, g2_bh, gh2, NN, 64, 192, 0);
  gru_final_kernel<<<(N64 + 255) / 256, 256, 0, stream>>>(gi2, gh2, x1o, out, NN, 64);
}
